// TimeSformerBlock_34892314313149
// MI455X (gfx1250) — matrix operand order repaired and checked by reading, not compiled, not run on hardware
//
#include <hip/hip_runtime.h>
#include <math.h>

// ---------------------------------------------------------------- types
typedef __bf16 bf16_t;
typedef bf16_t v16bf  __attribute__((ext_vector_type(16)));
typedef bf16_t bf16x8 __attribute__((ext_vector_type(8)));
typedef bf16_t bf16x2 __attribute__((ext_vector_type(2)));
typedef float  v8f    __attribute__((ext_vector_type(8)));
typedef float  f32x8  __attribute__((ext_vector_type(8)));
typedef float  f32x4  __attribute__((ext_vector_type(4)));
typedef float  f32x2  __attribute__((ext_vector_type(2)));
typedef unsigned short u16x8 __attribute__((ext_vector_type(8)));

static __device__ __forceinline__ unsigned short bf16_bits(float f) {
  union { bf16_t b; unsigned short u; } c; c.b = (bf16_t)f; return c.u;
}
// pairwise f32->bf16 pack: matches v_cvt_pk_bf16_f32 dst, a, b
static __device__ __forceinline__ unsigned pack_bf16(float lo, float hi) {
  f32x2 f = {lo, hi};
  union { bf16x2 b; unsigned u; } c;
  c.b = __builtin_convertvector(f, bf16x2);
  return c.u;
}
static __device__ __forceinline__ v8f zero8() {
  v8f z = {0.f,0.f,0.f,0.f,0.f,0.f,0.f,0.f}; return z;
}
static __device__ __forceinline__ v8f wmma_bf16(v16bf a, v16bf b, v8f c) {
  return __builtin_amdgcn_wmma_f32_16x16x32_bf16(false, a, false, b, (short)0, c,
                                                 false, false);
}

// A-fragment (16x32 bf16): elements 0..7 -> K = half*8+0..7, 8..15 -> 16+half*8+0..7
static __device__ __forceinline__ v16bf loadA_lds(const unsigned short* row, int half) {
  union { u16x8 v[2]; v16bf b; } u;
  u.v[0] = *(const u16x8*)(row + half * 8);
  u.v[1] = *(const u16x8*)(row + 16 + half * 8);
  return u.b;
}
// B-fragment (32x16 bf16): elements 0..7 -> K = half*8+0..7, 8..15 -> 16+half*8+0..7
static __device__ __forceinline__ v16bf loadB_lds(const unsigned short* row, int half) {
  union { u16x8 v[2]; v16bf b; } u;
  u.v[0] = *(const u16x8*)(row + half * 8);
  u.v[1] = *(const u16x8*)(row + 16 + half * 8);
  return u.b;
}
// Same runs straight from contiguous f32 global memory (b128 loads + paired cvt)
static __device__ __forceinline__ v16bf loadA_g(const float* row, int half, float scale) {
  f32x8 lo = *(const f32x8*)(row + half * 8);
  f32x8 hi = *(const f32x8*)(row + 16 + half * 8);
  union { unsigned w[8]; v16bf v; } u;
#pragma unroll
  for (int q = 0; q < 4; ++q) u.w[q]     = pack_bf16(lo[2*q] * scale, lo[2*q+1] * scale);
#pragma unroll
  for (int q = 0; q < 4; ++q) u.w[4 + q] = pack_bf16(hi[2*q] * scale, hi[2*q+1] * scale);
  return u.v;
}
static __device__ __forceinline__ v16bf loadB_g(const float* row, int half) {
  f32x8 lo = *(const f32x8*)(row + half * 8);
  f32x8 hi = *(const f32x8*)(row + 16 + half * 8);
  union { unsigned w[8]; v16bf v; } u;
#pragma unroll
  for (int q = 0; q < 4; ++q) u.w[q]     = pack_bf16(lo[2*q], lo[2*q+1]);
#pragma unroll
  for (int q = 0; q < 4; ++q) u.w[4 + q] = pack_bf16(hi[2*q], hi[2*q+1]);
  return u.v;
}

// ---------------------------------------------------------------- constants
#define FRAMES  256
#define PATCHES 64
#define DIM     512
#define HEADS   8
#define DH      64
#define DFF     1024
#define SEQ_T   257
#define SEQ_S   65
#define LK      128
#define QKVW    1536

// ---------------------------------------------------------------- gathers
__global__ void build_xt_kernel(const float* __restrict__ x, float* __restrict__ xt) {
  size_t i = (size_t)blockIdx.x * blockDim.x + threadIdx.x;
  const size_t total = (size_t)PATCHES * SEQ_T * DIM;
  if (i >= total) return;
  int d = (int)(i & (DIM - 1));
  size_t pt = i >> 9;
  int t = (int)(pt % SEQ_T);
  int p = (int)(pt / SEQ_T);
  size_t src = (t == 0) ? (size_t)d
                        : ((size_t)(1 + (size_t)(t - 1) * PATCHES + p) * DIM + d);
  xt[i] = x[src];
}

__global__ void build_ys_kernel(const float* __restrict__ yt, float* __restrict__ ys) {
  size_t i = (size_t)blockIdx.x * blockDim.x + threadIdx.x;
  const size_t total = (size_t)FRAMES * SEQ_S * DIM;
  if (i >= total) return;
  int d = (int)(i & (DIM - 1));
  size_t fs = i >> 9;
  int s = (int)(fs % SEQ_S);
  int f = (int)(fs / SEQ_S);
  size_t srcrow = (s == 0) ? ((size_t)(f & (PATCHES - 1)) * SEQ_T)
                           : ((size_t)(s - 1) * SEQ_T + (1 + f));
  ys[i] = yt[srcrow * DIM + d];
}

// ---------------------------------------------------------------- weight prep
// WT[n][k] = bf16(W[k][n]) : one-time f32 -> bf16 transpose.
__global__ __launch_bounds__(256) void transpose_w_bf16(
    const float* __restrict__ W, unsigned short* __restrict__ WT, int K, int N) {
  int idx = blockIdx.x * 256 + threadIdx.x;
  int kq = K >> 3;
  if (idx >= N * kq) return;
  int n = idx / kq;
  int k8 = (idx - n * kq) << 3;
  u16x8 o;
#pragma unroll
  for (int e = 0; e < 8; ++e) o[e] = bf16_bits(W[(size_t)(k8 + e) * N + n]);
  *(u16x8*)(WT + (size_t)n * K + k8) = o;
}

// ---------------------------------------------------------------- GEMM (bf16 WMMA)
// C = act(A*BT^T + bias) + resid. A: f32 MxK. BT: bf16 NxK (pre-transposed).
// Block 128x128, BK=32, 8 waves (4x2), 32x64 per wave -> 8 acc tiles.
// Out-of-range rows/cols are CLAMPED (not zeroed): their products only reach
// output elements the epilogue never stores, so staging is branchless.
// B tile is staged with GLOBAL_LOAD_ASYNC_TO_LDS_B128 (ASYNCcnt, no VGPR bounce).
#define BM 128
#define BN 128
#define BKK 32
#define ASTR 40   // u16 row stride: 80B, 16B aligned
#define BSTR 40

__global__ __launch_bounds__(256) void gemm_bf16wmma(
    const float* __restrict__ A, const unsigned short* __restrict__ BT,
    float* __restrict__ C, int M, int N, int K,
    const float* __restrict__ bias, const float* __restrict__ resid, int act) {
  __shared__ alignas(16) unsigned short As[BM * ASTR];
  __shared__ alignas(16) unsigned short Bs[BN * BSTR];
  const int tid  = threadIdx.x;
  const int lane = tid & 31, lid = lane & 15, half = lane >> 4;
  const int w = tid >> 5;
  const int wr = w & 3, wc = w >> 2;
  const int bm = blockIdx.y * BM;
  const int bn = blockIdx.x * BN;

  v8f acc[2][4];
#pragma unroll
  for (int i = 0; i < 2; ++i)
#pragma unroll
    for (int j = 0; j < 4; ++j) acc[i][j] = zero8();

  for (int k0 = 0; k0 < K; k0 += BKK) {
    // stage A: 8 contiguous f32 -> 4 packed cvt -> one ds_store_b128 per item
#pragma unroll
    for (int i = tid; i < BM * 4; i += 256) {
      int r = i >> 2, seg = i & 3;
      int gr = bm + r; if (gr >= M) gr = M - 1;           // clamp, no branch
      f32x8 v = *(const f32x8*)(A + (size_t)gr * K + k0 + seg * 8);
      union { unsigned w[4]; u16x8 u; } cv;
#pragma unroll
      for (int q = 0; q < 4; ++q) cv.w[q] = pack_bf16(v[2*q], v[2*q+1]);
      *(u16x8*)(As + r * ASTR + seg * 8) = cv.u;
    }
    // stage B: async global->LDS b128 copies of pre-transposed bf16 weights
#pragma unroll
    for (int i = tid; i < BN * 4; i += 256) {
      int n = i >> 2, seg = i & 3;
      int gn = bn + n; if (gn >= N) gn = N - 1;           // clamp, no branch
      const unsigned short* src = BT + (size_t)gn * K + k0 + seg * 8;
      unsigned dst = (unsigned)(unsigned long long)(uintptr_t)(Bs + n * BSTR + seg * 8);
      asm volatile("global_load_async_to_lds_b128 %0, %1, off"
                   :: "v"(dst), "v"(src) : "memory");
    }
    asm volatile("s_wait_asynccnt 0x0" ::: "memory");
    __syncthreads();

    v16bf af[2], bfr[4];
#pragma unroll
    for (int i = 0; i < 2; ++i)
      af[i] = loadA_lds(As + (wr * 32 + i * 16 + lid) * ASTR, half);
#pragma unroll
    for (int j = 0; j < 4; ++j)
      bfr[j] = loadB_lds(Bs + (wc * 64 + j * 16 + lid) * BSTR, half);
#pragma unroll
    for (int i = 0; i < 2; ++i)
#pragma unroll
      for (int j = 0; j < 4; ++j) acc[i][j] = wmma_bf16(af[i], bfr[j], acc[i][j]);
    __syncthreads();
  }

#pragma unroll
  for (int i = 0; i < 2; ++i)
#pragma unroll
    for (int j = 0; j < 4; ++j) {
      int n = bn + wc * 64 + j * 16 + lid;
      if (n >= N) continue;
      float bv = bias ? bias[n] : 0.f;
#pragma unroll
      for (int r = 0; r < 8; ++r) {
        int m = bm + wr * 32 + i * 16 + half * 8 + r;
        if (m >= M) continue;
        float v = acc[i][j][r] + bv;
        if (act) v = 0.5f * v * (1.f + erff(v * 0.70710678118f));  // exact GELU
        if (resid) v += resid[(size_t)m * N + n];
        C[(size_t)m * N + n] = v;
      }
    }
}

// ---------------------------------------------------------------- temporal MHSA
#define KT_TILES 17
#define KT_PAD   288     // 9 * 32

__global__ __launch_bounds__(32) void attn_temporal(
    const float* __restrict__ qkv, float* __restrict__ out) {
  __shared__ alignas(16) float          S[16 * KT_PAD];
  __shared__ alignas(16) unsigned short P[16 * KT_PAD];
  __shared__ alignas(16) unsigned short VT[64 * KT_PAD];    // VT[d][j]
  const int bh = blockIdx.x;
  const int p = bh >> 3, h = bh & 7;
  const int qbase = blockIdx.y * 16;
  const int lane = threadIdx.x, lid = lane & 15, half = lane >> 4;
  const float* base = qkv + (size_t)p * SEQ_T * QKVW + h * DH;
  const float* Qb = base;
  const float* Kb = base + DIM;
  const float* Vb = base + 2 * DIM;

  // stage V transposed: valid region (branchless) + zero pad
  {
    unsigned short* vt0 = VT + (2 * lane) * KT_PAD;
    unsigned short* vt1 = vt0 + KT_PAD;
#pragma clang loop unroll_count(2)
    for (int j = 0; j < SEQ_T; ++j) {
      f32x2 v = *(const f32x2*)(Vb + (size_t)j * QKVW + 2 * lane);
      unsigned pk = pack_bf16(v.x, v.y);
      vt0[j] = (unsigned short)pk;
      vt1[j] = (unsigned short)(pk >> 16);
    }
#pragma clang loop unroll(disable)
    for (int j = SEQ_T; j < KT_PAD; ++j) { vt0[j] = 0; vt1[j] = 0; }
  }

  int qrow = qbase + lid; if (qrow >= SEQ_T) qrow = SEQ_T - 1;
  const float* qrp = Qb + (size_t)qrow * QKVW;
  v16bf aQ0 = loadA_g(qrp, half, 0.125f);
  v16bf aQ1 = loadA_g(qrp + 32, half, 0.125f);

  for (int kt = 0; kt < KT_TILES; ++kt) {
    int krow = kt * 16 + lid; if (krow >= SEQ_T) krow = SEQ_T - 1;
    const float* krp = Kb + (size_t)krow * QKVW;
    v8f s = zero8();
    s = wmma_bf16(aQ0, loadB_g(krp, half), s);
    s = wmma_bf16(aQ1, loadB_g(krp + 32, half), s);
#pragma unroll
    for (int r = 0; r < 8; ++r) S[(half * 8 + r) * KT_PAD + kt * 16 + lid] = s[r];
  }

  if (lane < 16) {
    float mx = -1e30f;
#pragma clang loop unroll_count(4)
    for (int j = 0; j < SEQ_T; ++j) mx = fmaxf(mx, S[lane * KT_PAD + j]);
    float sum = 0.f;
#pragma clang loop unroll_count(4)
    for (int j = 0; j < SEQ_T; ++j) {
      float e = __expf(S[lane * KT_PAD + j] - mx);
      S[lane * KT_PAD + j] = e; sum += e;
    }
    float inv = 1.f / sum;
#pragma clang loop unroll_count(4)
    for (int j = 0; j < SEQ_T; ++j)
      P[lane * KT_PAD + j] = bf16_bits(S[lane * KT_PAD + j] * inv);
#pragma clang loop unroll(disable)
    for (int j = SEQ_T; j < KT_PAD; ++j) P[lane * KT_PAD + j] = 0;
  }

#pragma unroll
  for (int dt = 0; dt < 4; ++dt) {
    v8f o = zero8();
    for (int kc = 0; kc < 9; ++kc)
      o = wmma_bf16(loadA_lds(P + lid * KT_PAD + kc * 32, half),
                    loadB_lds(VT + (dt * 16 + lid) * KT_PAD + kc * 32, half), o);
#pragma unroll
    for (int r = 0; r < 8; ++r) {
      int qr = qbase + half * 8 + r;
      if (qr < SEQ_T)
        out[((size_t)p * SEQ_T + qr) * DIM + h * DH + dt * 16 + lid] = o[r];
    }
  }
}

// ---------------------------------------------------------------- spatial Linformer
#define JP 96   // padded contraction length (65 -> 3 chunks of 32)

__global__ __launch_bounds__(32) void attn_spatial(
    const float* __restrict__ qkv, const float* __restrict__ Emat,
    float* __restrict__ out) {
  __shared__ alignas(16) unsigned char smem[81920];
  unsigned short* ET  = (unsigned short*)(smem);            // [128][96]
  unsigned short* KT  = (unsigned short*)(smem + 24576);    // [64][96]
  unsigned short* VT  = (unsigned short*)(smem + 36864);    // [64][96]
  unsigned short* Kp  = (unsigned short*)(smem + 49152);    // [128][64]
  unsigned short* VpT = (unsigned short*)(smem + 65536);    // [64][128]
  float*          S   = (float*)(smem);                     // [16][128] (aliases ET)
  unsigned short* Pp  = (unsigned short*)(smem + 8192);     // [16][128]

  const int fh = blockIdx.x;
  const int f = fh >> 3, h = fh & 7;
  const int lane = threadIdx.x, lid = lane & 15, half = lane >> 4;
  const float* base = qkv + (size_t)f * SEQ_S * QKVW + h * DH;
  const float* Qb = base;
  const float* Kb = base + DIM;
  const float* Vb = base + 2 * DIM;

  // ---- phase A: stage E^T, K^T, V^T (valid region branchless, then pad)
  {
    unsigned short* e0 = ET + (4 * lane) * JP;
#pragma clang loop unroll_count(2)
    for (int j = 0; j < SEQ_S; ++j) {
      f32x4 v = *(const f32x4*)(Emat + (size_t)j * LK + 4 * lane);
      e0[j]          = bf16_bits(v.x);
      e0[JP + j]     = bf16_bits(v.y);
      e0[2 * JP + j] = bf16_bits(v.z);
      e0[3 * JP + j] = bf16_bits(v.w);
    }
#pragma clang loop unroll(disable)
    for (int j = SEQ_S; j < JP; ++j) {
      e0[j] = 0; e0[JP + j] = 0; e0[2 * JP + j] = 0; e0[3 * JP + j] = 0;
    }
  }
  {
    unsigned short* kt0 = KT + (2 * lane) * JP;
    unsigned short* vt0 = VT + (2 * lane) * JP;
#pragma clang loop unroll_count(2)
    for (int j = 0; j < SEQ_S; ++j) {
      f32x2 kv = *(const f32x2*)(Kb + (size_t)j * QKVW + 2 * lane);
      f32x2 vv = *(const f32x2*)(Vb + (size_t)j * QKVW + 2 * lane);
      unsigned pk = pack_bf16(kv.x, kv.y);
      unsigned pv = pack_bf16(vv.x, vv.y);
      kt0[j]      = (unsigned short)pk;
      kt0[JP + j] = (unsigned short)(pk >> 16);
      vt0[j]      = (unsigned short)pv;
      vt0[JP + j] = (unsigned short)(pv >> 16);
    }
#pragma clang loop unroll(disable)
    for (int j = SEQ_S; j < JP; ++j) {
      kt0[j] = 0; kt0[JP + j] = 0; vt0[j] = 0; vt0[JP + j] = 0;
    }
  }

  // ---- phase B: Kp = E^T x K ; VpT = V^T x E
  for (int mt = 0; mt < 8; ++mt)
#pragma unroll
    for (int dt = 0; dt < 4; ++dt) {
      v8f accv = zero8();
#pragma unroll
      for (int jc = 0; jc < 3; ++jc)
        accv = wmma_bf16(loadA_lds(ET + (mt * 16 + lid) * JP + jc * 32, half),
                         loadB_lds(KT + (dt * 16 + lid) * JP + jc * 32, half), accv);
#pragma unroll
      for (int r = 0; r < 8; ++r)
        Kp[(mt * 16 + half * 8 + r) * DH + dt * 16 + lid] = bf16_bits(accv[r]);
    }
  for (int mt = 0; mt < 4; ++mt)
#pragma unroll
    for (int nt = 0; nt < 8; ++nt) {
      v8f accv = zero8();
#pragma unroll
      for (int jc = 0; jc < 3; ++jc)
        accv = wmma_bf16(loadA_lds(VT + (mt * 16 + lid) * JP + jc * 32, half),
                         loadB_lds(ET + (nt * 16 + lid) * JP + jc * 32, half), accv);
#pragma unroll
      for (int r = 0; r < 8; ++r)
        VpT[(mt * 16 + half * 8 + r) * LK + nt * 16 + lid] = bf16_bits(accv[r]);
    }

  // ---- phase C: per q-tile scores / softmax / AV
  for (int qt = 0; qt < 5; ++qt) {
    int qrow = qt * 16 + lid; if (qrow >= SEQ_S) qrow = SEQ_S - 1;
    const float* qrp = Qb + (size_t)qrow * QKVW;
    v16bf aQ0 = loadA_g(qrp, half, 0.125f);
    v16bf aQ1 = loadA_g(qrp + 32, half, 0.125f);

#pragma unroll
    for (int nt = 0; nt < 8; ++nt) {
      const unsigned short* kprow = Kp + (nt * 16 + lid) * DH;
      v8f s = zero8();
      s = wmma_bf16(aQ0, loadB_lds(kprow, half), s);
      s = wmma_bf16(aQ1, loadB_lds(kprow + 32, half), s);
#pragma unroll
      for (int r = 0; r < 8; ++r) S[(half * 8 + r) * LK + nt * 16 + lid] = s[r];
    }

    if (lane < 16) {
      float mx = -1e30f;
#pragma clang loop unroll_count(4)
      for (int j = 0; j < LK; ++j) mx = fmaxf(mx, S[lane * LK + j]);
      float sum = 0.f;
#pragma clang loop unroll_count(4)
      for (int j = 0; j < LK; ++j) {
        float e = __expf(S[lane * LK + j] - mx);
        S[lane * LK + j] = e; sum += e;
      }
      float inv = 1.f / sum;
#pragma clang loop unroll_count(4)
      for (int j = 0; j < LK; ++j) Pp[lane * LK + j] = bf16_bits(S[lane * LK + j] * inv);
    }

#pragma unroll
    for (int dt = 0; dt < 4; ++dt) {
      v8f o = zero8();
#pragma unroll
      for (int kc = 0; kc < 4; ++kc)
        o = wmma_bf16(loadA_lds(Pp + lid * LK + kc * 32, half),
                      loadB_lds(VpT + (dt * 16 + lid) * LK + kc * 32, half), o);
#pragma unroll
      for (int r = 0; r < 8; ++r) {
        int qr = qt * 16 + half * 8 + r;
        if (qr < SEQ_S)
          out[((size_t)f * SEQ_S + qr) * DIM + h * DH + dt * 16 + lid] = o[r];
      }
    }
  }
}

// ---------------------------------------------------------------- launch
extern "C" void kernel_launch(void* const* d_in, const int* in_sizes, int n_in,
                              void* d_out, int out_size, void* d_ws, size_t ws_size,
                              hipStream_t stream) {
  (void)in_sizes; (void)n_in; (void)out_size; (void)ws_size;
  const float* x      = (const float*)d_in[0];
  const float* Wqkv_t = (const float*)d_in[1];
  const float* Wo_t   = (const float*)d_in[2];
  const float* Wqkv_s = (const float*)d_in[3];
  const float* Wo_s   = (const float*)d_in[4];
  const float* Emat   = (const float*)d_in[5];
  const float* W1     = (const float*)d_in[6];
  const float* b1     = (const float*)d_in[7];
  const float* W2     = (const float*)d_in[8];
  const float* b2     = (const float*)d_in[9];
  float* outp = (float*)d_out;

  const size_t N_T = (size_t)PATCHES * SEQ_T;   // 16448
  const size_t N_S = (size_t)FRAMES * SEQ_S;    // 16640

  float* ws   = (float*)d_ws;
  float* regA = ws;                       // xt -> y2        : 8,519,680 f32
  float* regB = regA + 8519680ULL;        // qkv_t / qkv_s   : 25,559,040
  float* regC = regB + 25559040ULL;       // attn out        : 8,519,680
  float* regD = regC + 8519680ULL;        // y_t -> ffn h    : 17,039,360
  float* regE = regD + 17039360ULL;       // ys              : 8,519,680

  float* xt   = regA;  float* y2  = regA;
  float* qkvb = regB;
  float* attn = regC;
  float* yt   = regD;  float* ffn = regD;
  float* ys   = regE;

  unsigned short* wsbf    = (unsigned short*)(regE + 8519680ULL);
  unsigned short* WqkvT_t = wsbf;                       // 1536*512
  unsigned short* WoT_t   = WqkvT_t + 786432ULL;        //  512*512
  unsigned short* WqkvT_s = WoT_t   + 262144ULL;        // 1536*512
  unsigned short* WoT_s   = WqkvT_s + 786432ULL;        //  512*512
  unsigned short* W1T     = WoT_s   + 262144ULL;        // 1024*512
  unsigned short* W2T     = W1T     + 524288ULL;        //  512*1024

  transpose_w_bf16<<<384, 256, 0, stream>>>(Wqkv_t, WqkvT_t, DIM, QKVW);
  transpose_w_bf16<<<128, 256, 0, stream>>>(Wo_t,   WoT_t,   DIM, DIM);
  transpose_w_bf16<<<384, 256, 0, stream>>>(Wqkv_s, WqkvT_s, DIM, QKVW);
  transpose_w_bf16<<<128, 256, 0, stream>>>(Wo_s,   WoT_s,   DIM, DIM);
  transpose_w_bf16<<<256, 256, 0, stream>>>(W1,     W1T,     DIM, DFF);
  transpose_w_bf16<<<256, 256, 0, stream>>>(W2,     W2T,     DFF, DIM);

  build_xt_kernel<<<(unsigned)((N_T * DIM) / 256), 256, 0, stream>>>(x, xt);
  gemm_bf16wmma<<<dim3(QKVW / BN, (unsigned)((N_T + BM - 1) / BM)), 256, 0, stream>>>(
      xt, WqkvT_t, qkvb, (int)N_T, QKVW, DIM, nullptr, nullptr, 0);
  attn_temporal<<<dim3(PATCHES * HEADS, KT_TILES), 32, 0, stream>>>(qkvb, attn);
  gemm_bf16wmma<<<dim3(DIM / BN, (unsigned)((N_T + BM - 1) / BM)), 256, 0, stream>>>(
      attn, WoT_t, yt, (int)N_T, DIM, DIM, nullptr, xt, 0);
  build_ys_kernel<<<(unsigned)((N_S * DIM) / 256), 256, 0, stream>>>(yt, ys);
  gemm_bf16wmma<<<dim3(QKVW / BN, (unsigned)((N_S + BM - 1) / BM)), 256, 0, stream>>>(
      ys, WqkvT_s, qkvb, (int)N_S, QKVW, DIM, nullptr, nullptr, 0);
  attn_spatial<<<dim3(FRAMES * HEADS), 32, 0, stream>>>(qkvb, Emat, attn);
  gemm_bf16wmma<<<dim3(DIM / BN, (unsigned)((N_S + BM - 1) / BM)), 256, 0, stream>>>(
      attn, WoT_s, y2, (int)N_S, DIM, DIM, nullptr, ys, 0);
  gemm_bf16wmma<<<dim3(DFF / BN, (unsigned)((N_S + BM - 1) / BM)), 256, 0, stream>>>(
      y2, W1T, ffn, (int)N_S, DFF, DIM, b1, nullptr, 1);
  gemm_bf16wmma<<<dim3(DIM / BN, (unsigned)((N_S + BM - 1) / BM)), 256, 0, stream>>>(
      ffn, W2T, outp, (int)N_S, DIM, DFF, b2, y2, 0);
}
